// GraphAttention_79422535238377
// MI455X (gfx1250) — matrix operand order repaired and checked by reading, not compiled, not run on hardware
//
#include <hip/hip_runtime.h>
#include <hip/hip_bf16.h>

typedef __attribute__((ext_vector_type(16))) _Float16 v16h;
typedef __attribute__((ext_vector_type(8)))  float    v8f;

#define F_DIM 128
#define NEG_SLOPE 0.01f
#define EPS 1e-12f

// ---------------------------------------------------------------------------
// Kernel 0: zero the accumulation buffers (d_out rows + denom) each call so
// graph replays are deterministic.
// ---------------------------------------------------------------------------
__global__ void gat_zero(float* __restrict__ out, float* __restrict__ denom, int N) {
    int i = blockIdx.x * blockDim.x + threadIdx.x;
    int total = N * F_DIM;
    if (i < total) out[i] = 0.0f;
    if (i < N)     denom[i] = 0.0f;
}

// ---------------------------------------------------------------------------
// Kernel 1: messages = x @ W^T via V_WMMA_F32_16X16X32_F16.
// One wave per 16x16 output tile; K=128 -> 4 WMMA ops per tile.
// A-fragment layout (16-bit A 16x32): lane l and lane l+16 both hold row M=l;
//   half-wave g=0 covers K {0..7,16..23}, g=1 covers K {8..15,24..31};
//   VGPR j (element pair 2j,2j+1): k = (j>=4?16:0) + (j&3)*2 + g*8 + {0,1}.
// B-fragment layout (32x16): N = lane&15; same K mapping as A:
//   VGPR j packs K = (j>=4?16:0) + (j&3)*2 + g*8 + {0,1}.
// C layout: VGPR r -> row m0 + r + g*8, col n0 + (lane&15).
// ---------------------------------------------------------------------------
__global__ void gat_gemm_wmma(const float* __restrict__ x,
                              const float* __restrict__ W,
                              float* __restrict__ msg, int N) {
    int wave = (blockIdx.x * blockDim.x + threadIdx.x) >> 5;
    int lane = threadIdx.x & 31;
    int mTile = wave >> 3;          // F_DIM/16 = 8 column tiles
    int nTile = wave & 7;
    int m0 = mTile * 16;
    if (m0 >= N) return;
    int n0 = nTile * 16;
    int g = lane >> 4;              // half-wave group
    int l = lane & 15;

    int arow = m0 + l;  if (arow >= N) arow = N - 1;   // tail guard (N%16==0 normally)
    const float* __restrict__ xr = x + (size_t)arow * F_DIM;
    const float* __restrict__ wr = W + (size_t)(n0 + l) * F_DIM;

    v8f acc = {};
    #pragma unroll
    for (int k0 = 0; k0 < F_DIM; k0 += 32) {
        v16h A, B;
        #pragma unroll
        for (int j = 0; j < 8; ++j) {
            int ka = k0 + ((j >= 4) ? 16 : 0) + ((j & 3) << 1) + (g << 3);
            A[2*j]   = (_Float16)xr[ka];
            A[2*j+1] = (_Float16)xr[ka + 1];
            int kb = k0 + ((j >= 4) ? 16 : 0) + ((j & 3) << 1) + (g << 3);
            B[2*j]   = (_Float16)wr[kb];
            B[2*j+1] = (_Float16)wr[kb + 1];
        }
        acc = __builtin_amdgcn_wmma_f32_16x16x32_f16(
            /*neg_a=*/false, A, /*neg_b=*/false, B,
            /*c_mod=*/(short)0, acc, /*reuse_a=*/false, /*reuse_b=*/false);
    }

    #pragma unroll
    for (int r = 0; r < 8; ++r) {
        int row = m0 + r + (g << 3);
        if (row < N) msg[(size_t)row * F_DIM + n0 + l] = acc[r];
    }
}

// ---------------------------------------------------------------------------
// Kernel 2: s_src[n] = <messages[n], a[0:128]>, s_dst[n] = <messages[n], a[128:256]>
// One wave per node, wave32 shuffle reduction.
// ---------------------------------------------------------------------------
__global__ void gat_scores(const float* __restrict__ msg, const float* __restrict__ a,
                           float* __restrict__ s_src, float* __restrict__ s_dst, int N) {
    int wave = (blockIdx.x * blockDim.x + threadIdx.x) >> 5;
    int lane = threadIdx.x & 31;
    if (wave >= N) return;
    const float* m = msg + (size_t)wave * F_DIM;
    float ss = 0.0f, sd = 0.0f;
    #pragma unroll
    for (int k = lane; k < F_DIM; k += 32) {
        float v = m[k];
        ss += v * a[k];
        sd += v * a[F_DIM + k];
    }
    #pragma unroll
    for (int off = 16; off > 0; off >>= 1) {
        ss += __shfl_xor(ss, off, 32);
        sd += __shfl_xor(sd, off, 32);
    }
    if (lane == 0) { s_src[wave] = ss; s_dst[wave] = sd; }
}

// ---------------------------------------------------------------------------
// Kernel 3: per-edge unnormalized weight w = exp(leaky_relu(s_src[src]+s_dst[dst]))
// and denom scatter-add. Edges [0,E) real, [E,E+N) self loops.
// ---------------------------------------------------------------------------
__global__ void gat_edge_w(const int* __restrict__ ei,
                           const float* __restrict__ s_src, const float* __restrict__ s_dst,
                           float* __restrict__ w, float* __restrict__ denom, int E, int N) {
    int e = blockIdx.x * blockDim.x + threadIdx.x;
    int total = E + N;
    if (e >= total) return;
    int src, dst;
    if (e < E) { src = ei[e]; dst = ei[E + e]; }
    else       { src = dst = e - E; }
    float sc = s_src[src] + s_dst[dst];
    float lr = (sc > 0.0f) ? sc : NEG_SLOPE * sc;
    float wv = __expf(lr);
    w[e] = wv;
    atomicAdd(&denom[dst], wv);
}

// ---------------------------------------------------------------------------
// Kernel 4: feature aggregation. One wave per edge: lanes gather a coalesced
// 128-float message row (L2-resident) and scatter w*msg into out[dst] with
// f32 atomics.
// ---------------------------------------------------------------------------
__global__ void gat_aggregate(const int* __restrict__ ei, const float* __restrict__ w,
                              const float* __restrict__ msg, float* __restrict__ out,
                              int E, int N) {
    int wave = (blockIdx.x * blockDim.x + threadIdx.x) >> 5;
    int lane = threadIdx.x & 31;
    int total = E + N;
    if (wave >= total) return;
    int src, dst;
    if (wave < E) { src = ei[wave]; dst = ei[E + wave]; }
    else          { src = dst = wave - E; }
    float wv = w[wave];
    const float* m = msg + (size_t)src * F_DIM;
    float*       o = out + (size_t)dst * F_DIM;
    #pragma unroll
    for (int f = lane; f < F_DIM; f += 32)
        atomicAdd(&o[f], wv * m[f]);
}

// ---------------------------------------------------------------------------
// Kernel 5: normalize: out[n,:] /= max(denom[n], EPS)
// ---------------------------------------------------------------------------
__global__ void gat_finalize(float* __restrict__ out, const float* __restrict__ denom, int N) {
    int i = blockIdx.x * blockDim.x + threadIdx.x;
    if (i >= N * F_DIM) return;
    float d = fmaxf(denom[i >> 7], EPS);
    out[i] = out[i] / d;
}

extern "C" void kernel_launch(void* const* d_in, const int* in_sizes, int n_in,
                              void* d_out, int out_size, void* d_ws, size_t ws_size,
                              hipStream_t stream) {
    const float* x  = (const float*)d_in[0];
    const int*   ei = (const int*)d_in[1];
    const float* W  = (const float*)d_in[2];
    const float* a  = (const float*)d_in[3];
    float* out = (float*)d_out;

    const int N = in_sizes[0] / F_DIM;   // 50000
    const int E = in_sizes[1] / 2;       // 600000

    // workspace layout (floats): msg[N*128] | s_src[N] | s_dst[N] | denom[N] | w[E+N]
    float* ws    = (float*)d_ws;
    float* msg   = ws;
    float* s_src = msg   + (size_t)N * F_DIM;
    float* s_dst = s_src + N;
    float* denom = s_dst + N;
    float* wbuf  = denom + N;

    const int totalOut = N * F_DIM;
    const int totalE   = E + N;

    gat_zero<<<(totalOut + 255) / 256, 256, 0, stream>>>(out, denom, N);

    int mTiles = (N + 15) / 16;
    int waves  = mTiles * (F_DIM / 16);            // 8 column tiles per row band
    gat_gemm_wmma<<<(waves + 7) / 8, 256, 0, stream>>>(x, W, msg, N);

    gat_scores<<<((size_t)N * 32 + 255) / 256, 256, 0, stream>>>(msg, a, s_src, s_dst, N);

    gat_edge_w<<<(totalE + 255) / 256, 256, 0, stream>>>(ei, s_src, s_dst, wbuf, denom, E, N);

    gat_aggregate<<<((size_t)totalE * 32 + 255) / 256, 256, 0, stream>>>(ei, wbuf, msg, out, E, N);

    gat_finalize<<<(totalOut + 255) / 256, 256, 0, stream>>>(out, denom, N);
}
